// MultiHeadSelfAttention_49813030699304
// MI455X (gfx1250) — matrix operand order repaired and checked by reading, not compiled, not run on hardware
//
#include <hip/hip_runtime.h>
#include <hip/hip_bf16.h>

#define BB 8
#define SS 1024
#define DD 384
#define HH 8
#define HDIM (HH * DD) /* 3072 */

typedef __attribute__((ext_vector_type(16))) __bf16       v16bf;
typedef __attribute__((ext_vector_type(8)))  float        v8f;
typedef __attribute__((ext_vector_type(4)))  unsigned int v4u;

union frag_u { v4u u[2]; v16bf v; };

__device__ __forceinline__ unsigned short f2bf(float f) {
  unsigned int u = __float_as_uint(f);
  u += 0x7FFFu + ((u >> 16) & 1u);
  return (unsigned short)(u >> 16);
}

__device__ __forceinline__ v8f wmma_bf16(v16bf a, v16bf b, v8f c) {
  // D = A(16x32 bf16) * B(32x16 bf16) + C(16x16 f32)
  return __builtin_amdgcn_wmma_f32_16x16x32_bf16(false, a, false, b, (short)0, c,
                                                 false, false);
}

// A fragment: rows row0..row0+15, K = col0..col0+31 of a row-major [*, ld] bf16 matrix.
// ISA 16-bit A layout: lanes 0-15 (row=lane) hold K 0-7 & 16-23; lanes 16-31 hold 8-15 & 24-31.
__device__ __forceinline__ v16bf load_a(const unsigned short* base, int row0, int col0, int ld) {
  const int lane = threadIdx.x & 31;
  const int m = lane & 15, sel = lane >> 4;
  const unsigned short* p = base + (long)(row0 + m) * ld + col0 + sel * 8;
  frag_u r;
  r.u[0] = *(const v4u*)(p);
  r.u[1] = *(const v4u*)(p + 16);
  return r.v;
}

// B fragment from K-major storage BT[N][ldt] (i.e. B transposed): covers B[k0..k0+31][n0..n0+15].
// ISA 16-bit B layout: lane n (0-15) holds K 0-7 & 16-23; lanes 16-31 hold K 8-15 & 24-31.
__device__ __forceinline__ v16bf load_b(const unsigned short* bt, int n0, int k0, int ldt) {
  const int lane = threadIdx.x & 31;
  const int nn = lane & 15, sel = lane >> 4;
  const unsigned short* p = bt + (long)(n0 + nn) * ldt + k0 + sel * 8;
  frag_u r;
  r.u[0] = *(const v4u*)(p);
  r.u[1] = *(const v4u*)(p + 16);
  return r.v;
}

// CDNA5 async memory->LDS copy, 16 bytes per lane, tracked by ASYNCcnt.
__device__ __forceinline__ void async_cp16(unsigned lds_byte, const unsigned short* g) {
  asm volatile("global_load_async_to_lds_b128 %0, %1, off"
               :: "v"(lds_byte), "v"(g)
               : "memory");
}

// ---------------- stage 1: conversion / transposition ----------------
__global__ void cvt_bf16_kernel(const float* __restrict__ src,
                                unsigned short* __restrict__ dst, int n) {
  int i = blockIdx.x * blockDim.x + threadIdx.x;
  int stride = gridDim.x * blockDim.x;
  for (; i < n; i += stride) dst[i] = f2bf(src[i]);
}

// wt[h][e][d] = w[h][d][e], converted to bf16 (K-major for WMMA B-fragments)
__global__ void transpose_w_kernel(const float* __restrict__ w,
                                   unsigned short* __restrict__ wt,
                                   int nh, int din, int dout) {
  long total = (long)nh * din * dout;
  long i = (long)blockIdx.x * blockDim.x + threadIdx.x;
  long stride = (long)gridDim.x * blockDim.x;
  for (; i < total; i += stride) {
    int e = (int)(i % dout);
    long t = i / dout;
    int d = (int)(t % din);
    int h = (int)(t / din);
    wt[((long)h * dout + e) * din + d] = f2bf(w[i]);
  }
}

// ---------------- stage 2: Q/K/V projections ----------------
__global__ void __launch_bounds__(256) qkv_proj_kernel(
    const unsigned short* __restrict__ xb,   // [B,S,D] bf16
    const unsigned short* __restrict__ wt,   // [H,Dout,Din] bf16 (K-major)
    const float* __restrict__ bias,          // [H,D]
    unsigned short* __restrict__ out, int transposed) {
  const int tid = threadIdx.x;
  const int wave = tid >> 5, lane = tid & 31;
  const int nn = lane & 15, half = lane >> 4;

  int bid = blockIdx.x;                 // 3 * 16 * 64 blocks
  const int nblk = bid % 3; bid /= 3;
  const int sblk = bid & 15; bid >>= 4;
  const int b = bid >> 3, h = bid & 7;

  const int mw = wave >> 1, nw = wave & 1;
  const int s0 = sblk * 64 + mw * 16;
  const int e0 = nblk * 128 + nw * 64;

  const unsigned short* xrow = xb + (long)b * SS * DD;
  const unsigned short* wh = wt + (long)h * DD * DD;

  const v8f vz = {0.f, 0.f, 0.f, 0.f, 0.f, 0.f, 0.f, 0.f};
  v8f acc[4];
#pragma unroll
  for (int j = 0; j < 4; j++) acc[j] = vz;

  for (int kk = 0; kk < DD; kk += 32) {
    v16bf a = load_a(xrow, s0, kk, DD);
#pragma unroll
    for (int j = 0; j < 4; j++) {
      v16bf bf = load_b(wh, e0 + j * 16, kk, DD);
      acc[j] = wmma_bf16(a, bf, acc[j]);
    }
  }

  if (!transposed) {
    unsigned short* ob = out + ((long)(b * HH + h)) * SS * DD;
#pragma unroll
    for (int j = 0; j < 4; j++) {
      const int e = e0 + j * 16 + nn;
      const float bv = bias[h * DD + e];
#pragma unroll
      for (int r = 0; r < 8; r++) {
        const int m = r + 8 * half;   // C layout: VGPR r, halves are rows r / r+8
        ob[(long)(s0 + m) * DD + e] = f2bf(acc[j][r] + bv);
      }
    }
  } else {
    unsigned short* ob = out + ((long)(b * HH + h)) * DD * SS;
#pragma unroll
    for (int j = 0; j < 4; j++) {
      const int e = e0 + j * 16 + nn;
      const float bv = bias[h * DD + e];
      v4u pk;
#pragma unroll
      for (int r = 0; r < 4; r++) {
        unsigned int lo = f2bf(acc[j][2 * r] + bv);
        unsigned int hi = f2bf(acc[j][2 * r + 1] + bv);
        pk[r] = lo | (hi << 16);
      }
      // transposed store: lane-contiguous 8 bf16 along S
      *(v4u*)(ob + (long)e * SS + s0 + 8 * half) = pk;
    }
  }
}

// ---------------- stage 3: flash attention ----------------
// Block = 8 waves: 4 query sub-tiles x 2 head-dim halves. K/V tiles for each
// 32-key chunk are staged into double-buffered LDS with async loads and shared
// by all waves; next chunk's DMA overlaps current chunk's WMMA work.
__global__ void __launch_bounds__(256) flash_attn_kernel(
    const unsigned short* __restrict__ Qg,   // [B,H,S,D]
    const unsigned short* __restrict__ Kg,   // [B,H,S,D]
    const unsigned short* __restrict__ VTg,  // [B,H,D,S]
    const int* __restrict__ mask,            // [B,S]
    unsigned short* __restrict__ concat) {   // [B,S,H*D]
  __shared__ __align__(16) unsigned short lds_k[2][32 * DD];  // 2 x 24KB
  __shared__ __align__(16) unsigned short lds_v[2][DD * 32];  // 2 x 24KB
  __shared__ __align__(16) unsigned short lds_p[8][16 * 32];  // 8KB

  const int tid = threadIdx.x;
  const int wave = tid >> 5, lane = tid & 31;
  const int nn = lane & 15, half = lane >> 4;

  const int qblk = blockIdx.x & 15;
  const int bh = blockIdx.x >> 4;
  const int b = bh >> 3, h = bh & 7;

  const int qsub = wave >> 1;          // 4 query sub-tiles of 16 rows
  const int dhalf = wave & 1;          // head-dim halves (192 each)
  const int q0 = qblk * 64 + qsub * 16;
  const int dbase = dhalf * 192;

  const unsigned short* Qb = Qg + (long)bh * SS * DD;
  const unsigned short* Kb = Kg + (long)bh * SS * DD;
  const unsigned short* Vb = VTg + (long)bh * DD * SS;
  const int* mb = mask + b * SS;

  const v8f vz = {0.f, 0.f, 0.f, 0.f, 0.f, 0.f, 0.f, 0.f};
  v8f Oacc[12];
#pragma unroll
  for (int i = 0; i < 12; i++) Oacc[i] = vz;
  float m_run[8], l_run[8];
#pragma unroll
  for (int r = 0; r < 8; r++) { m_run[r] = -1e30f; l_run[r] = 0.f; }

  const float scale = 0.051031036307982884f;  // 1/sqrt(384)
  const int kend = q0 + 15;                   // this wave's causal horizon
  const int kblkend = qblk * 64 + 63;         // block-uniform horizon

  // Stage K[kt..kt+31][:] (contiguous 24KB) and V^T[:][kt..kt+31] (384 rows x 64B)
  // as 12 x 16B async transfers per thread.
  auto stage_kv = [&](int kt, int buf) {
    const unsigned short* gk = Kb + (long)kt * DD;
    const unsigned kb = (unsigned)(size_t)&lds_k[buf][0];  // low 32 bits = LDS offset
    const unsigned vbb = (unsigned)(size_t)&lds_v[buf][0];
#pragma unroll
    for (int i = 0; i < 6; i++) {
      const int idx = tid + i * 256;
      async_cp16(kb + idx * 16, gk + idx * 8);
    }
#pragma unroll
    for (int i = 0; i < 6; i++) {
      const int idx = tid + i * 256;
      const int row = idx >> 2, c = idx & 3;
      async_cp16(vbb + idx * 16, Vb + (long)row * SS + kt + c * 8);
    }
  };

  stage_kv(0, 0);

  for (int kt = 0; kt <= kblkend; kt += 32) {
    const int buf = (kt >> 5) & 1;
    const bool has_next = (kt + 32) <= kblkend;
    if (has_next) {
      stage_kv(kt + 32, buf ^ 1);  // overlap next chunk's DMA with this chunk's math
      asm volatile("s_wait_asynccnt 12" ::: "memory");  // chunk kt landed; 12 newer in flight
    } else {
      asm volatile("s_wait_asynccnt 0" ::: "memory");
    }
    __syncthreads();

    if (kt <= kend) {  // wave-uniform causal guard
      const unsigned short* lk = &lds_k[buf][0];
      const unsigned short* lv = &lds_v[buf][0];

      // ---- scores for 32 keys: two 16x16 tiles, B-frags from LDS ----
      v8f c0 = vz, c1 = vz;
      for (int kk = 0; kk < DD; kk += 32) {
        v16bf a = load_a(Qb, q0, kk, DD);
        v16bf b0 = load_b(lk, 0, kk, DD);
        v16bf b1 = load_b(lk, 16, kk, DD);
        c0 = wmma_bf16(a, b0, c0);
        c1 = wmma_bf16(a, b1, c1);
      }

      const int key0 = kt + nn, key1 = kt + 16 + nn;
      const int pm0 = mb[key0], pm1 = mb[key1];
      float p0a[8], p1a[8];
#pragma unroll
      for (int r = 0; r < 8; r++) {
        const int qrow = q0 + r + 8 * half;
        float s0 = (key0 <= qrow && pm0 != 0) ? c0[r] * scale : -1e30f;
        float s1 = (key1 <= qrow && pm1 != 0) ? c1[r] * scale : -1e30f;
        float mx = fmaxf(s0, s1);
        mx = fmaxf(mx, __shfl_xor(mx, 1, 32));
        mx = fmaxf(mx, __shfl_xor(mx, 2, 32));
        mx = fmaxf(mx, __shfl_xor(mx, 4, 32));
        mx = fmaxf(mx, __shfl_xor(mx, 8, 32));
        const float mnew = fmaxf(m_run[r], mx);
        const float corr = __expf(m_run[r] - mnew);
        float p0 = (s0 > -5e29f) ? __expf(s0 - mnew) : 0.f;
        float p1 = (s1 > -5e29f) ? __expf(s1 - mnew) : 0.f;
        float ps = p0 + p1;
        ps += __shfl_xor(ps, 1, 32);
        ps += __shfl_xor(ps, 2, 32);
        ps += __shfl_xor(ps, 4, 32);
        ps += __shfl_xor(ps, 8, 32);
        l_run[r] = l_run[r] * corr + ps;
        m_run[r] = mnew;
        p0a[r] = p0; p1a[r] = p1;
#pragma unroll
        for (int nt = 0; nt < 12; nt++) Oacc[nt][r] *= corr;
      }

      // ---- C-layout -> A-layout via per-wave LDS round-trip ----
      unsigned short* lp = &lds_p[wave][0];
#pragma unroll
      for (int r = 0; r < 8; r++) {
        const int m = r + 8 * half;
        lp[m * 32 + nn] = f2bf(p0a[r]);
        lp[m * 32 + 16 + nn] = f2bf(p1a[r]);
      }
      asm volatile("s_wait_dscnt 0" ::: "memory");
      frag_u pa;
      const unsigned short* ap = lp + (lane & 15) * 32 + (lane >> 4) * 8;
      pa.u[0] = *(const v4u*)(ap);
      pa.u[1] = *(const v4u*)(ap + 16);
      asm volatile("" ::: "memory");

      // ---- P(16x32) @ V(32x16 tiles) over 192 head-dims, B-frags from LDS ----
#pragma unroll
      for (int nt = 0; nt < 12; nt++) {
        v16bf bv = load_b(lv, dbase + nt * 16, 0, 32);
        Oacc[nt] = wmma_bf16(pa.v, bv, Oacc[nt]);
      }
      asm volatile("s_wait_dscnt 0" ::: "memory");
    }
    __syncthreads();  // all waves done reading buf before it gets restaged
  }

  // ---- finalize: normalize, store to concat [B,S,H*D] ----
#pragma unroll
  for (int r = 0; r < 8; r++)
    m_run[r] = (m_run[r] > -5e29f && l_run[r] > 0.f) ? (1.0f / l_run[r]) : 0.f;
#pragma unroll
  for (int nt = 0; nt < 12; nt++) {
#pragma unroll
    for (int r = 0; r < 8; r++) {
      const int m = r + 8 * half;
      const int col = h * DD + dbase + nt * 16 + nn;
      concat[((long)b * SS + (q0 + m)) * HDIM + col] = f2bf(Oacc[nt][r] * m_run[r]);
    }
  }
}

// ---------------- stage 4: output projection ----------------
__global__ void __launch_bounds__(256) out_proj_kernel(
    const unsigned short* __restrict__ cc,   // [B*S, H*D] bf16
    const unsigned short* __restrict__ wot,  // [D, H*D] bf16 (K-major Wo)
    const float* __restrict__ bo,            // [D]
    const int* __restrict__ mask,            // [B*S]
    float* __restrict__ out) {               // [B*S, D]
  const int tid = threadIdx.x;
  const int wave = tid >> 5, lane = tid & 31;
  const int nn = lane & 15, half = lane >> 4;

  int bid = blockIdx.x;                 // 3 * 128 blocks
  const int nblk = bid % 3;
  const int mblk = bid / 3;
  const int mw = wave >> 1, nw = wave & 1;
  const int row0 = mblk * 64 + mw * 16;
  const int e0 = nblk * 128 + nw * 64;

  const v8f vz = {0.f, 0.f, 0.f, 0.f, 0.f, 0.f, 0.f, 0.f};
  v8f acc[4];
#pragma unroll
  for (int j = 0; j < 4; j++) acc[j] = vz;

  for (int kk = 0; kk < HDIM; kk += 32) {
    v16bf a = load_a(cc, row0, kk, HDIM);
#pragma unroll
    for (int j = 0; j < 4; j++) {
      v16bf bf = load_b(wot, e0 + j * 16, kk, HDIM);
      acc[j] = wmma_bf16(a, bf, acc[j]);
    }
  }

#pragma unroll
  for (int j = 0; j < 4; j++) {
    const int e = e0 + j * 16 + nn;
    const float bv = bo[e];
#pragma unroll
    for (int r = 0; r < 8; r++) {
      const int row = row0 + r + 8 * half;
      const float mv = (mask[row] != 0) ? 1.0f : 0.0f;
      out[(long)row * DD + e] = (acc[j][r] + bv) * mv;
    }
  }
}

extern "C" void kernel_launch(void* const* d_in, const int* in_sizes, int n_in,
                              void* d_out, int out_size, void* d_ws, size_t ws_size,
                              hipStream_t stream) {
  (void)in_sizes; (void)n_in; (void)out_size; (void)ws_size;
  const float* x  = (const float*)d_in[0];
  const int* mask = (const int*)d_in[1];
  const float* Wq = (const float*)d_in[2];
  const float* bq = (const float*)d_in[3];
  const float* Wk = (const float*)d_in[4];
  const float* bk = (const float*)d_in[5];
  const float* Wv = (const float*)d_in[6];
  const float* bv = (const float*)d_in[7];
  const float* Wo = (const float*)d_in[8];
  const float* bo = (const float*)d_in[9];
  float* out = (float*)d_out;

  char* ws = (char*)d_ws;
  size_t off = 0;
  auto carve = [&](size_t bytes) -> void* {
    void* p = ws + off;
    off = (off + bytes + 255) & ~(size_t)255;
    return p;
  };
  unsigned short* xb  = (unsigned short*)carve((size_t)BB * SS * DD * 2);
  unsigned short* wqT = (unsigned short*)carve((size_t)HH * DD * DD * 2);
  unsigned short* wkT = (unsigned short*)carve((size_t)HH * DD * DD * 2);
  unsigned short* wvT = (unsigned short*)carve((size_t)HH * DD * DD * 2);
  unsigned short* woT = (unsigned short*)carve((size_t)DD * HDIM * 2);
  unsigned short* Qg  = (unsigned short*)carve((size_t)BB * HH * SS * DD * 2);
  unsigned short* Kg  = (unsigned short*)carve((size_t)BB * HH * SS * DD * 2);
  unsigned short* VTg = (unsigned short*)carve((size_t)BB * HH * DD * SS * 2);
  unsigned short* ccg = (unsigned short*)carve((size_t)BB * SS * HDIM * 2);

  cvt_bf16_kernel<<<1024, 256, 0, stream>>>(x, xb, BB * SS * DD);
  transpose_w_kernel<<<1024, 256, 0, stream>>>(Wq, wqT, HH, DD, DD);
  transpose_w_kernel<<<1024, 256, 0, stream>>>(Wk, wkT, HH, DD, DD);
  transpose_w_kernel<<<1024, 256, 0, stream>>>(Wv, wvT, HH, DD, DD);
  transpose_w_kernel<<<1024, 256, 0, stream>>>(Wo, woT, 1, HDIM, DD);

  qkv_proj_kernel<<<3 * 16 * BB * HH, 256, 0, stream>>>(xb, wqT, bq, Qg, 0);
  qkv_proj_kernel<<<3 * 16 * BB * HH, 256, 0, stream>>>(xb, wkT, bk, Kg, 0);
  qkv_proj_kernel<<<3 * 16 * BB * HH, 256, 0, stream>>>(xb, wvT, bv, VTg, 1);

  flash_attn_kernel<<<BB * HH * (SS / 64), 256, 0, stream>>>(Qg, Kg, VTg, mask, ccg);

  out_proj_kernel<<<(BB * SS / 64) * 3, 256, 0, stream>>>(ccg, woT, bo, mask, out);
}
